// MultiResolutionHashTimeDoubleEncoderDoubleNetwork_21320217658073
// MI455X (gfx1250) — matrix operand order repaired and checked by reading, not compiled, not run on hardware
//
#include <hip/hip_runtime.h>

// ---------------------------------------------------------------------------
// Fully-fused hash-grid encoder + MLP stack for MI455X (gfx1250, wave32).
// Block = 128 threads = 4 waves; 64 points per block; each wave owns a
// 16-point M-tile through every WMMA layer. Activations live in LDS as fp16,
// weights are staged per-layer into LDS as fp16 transposed (W^T[n][k]).
// Matrix math: v_wmma_f32_16x16x32_f16 (f32 accumulate).
// ---------------------------------------------------------------------------

typedef __attribute__((ext_vector_type(16))) _Float16 v16h;
typedef __attribute__((ext_vector_type(8)))  float    v8f;
typedef __attribute__((ext_vector_type(4)))  float    f4;
typedef __attribute__((ext_vector_type(2)))  float    f2;

#define HASH_T_MASK 0x7FFFFu            // T = 2^19
#define TBL_LEVEL_STRIDE (524288u * 2u) // T * F floats per level

__constant__ float c_res[16] = {
    16.f, 24.f, 36.f, 54.f, 81.f, 121.f, 182.f, 273.f,
    410.f, 615.f, 922.f, 1383.f, 2075.f, 3113.f, 4670.f, 7006.f};

// ---- WMMA fragment helpers (layouts per CDNA5 ISA 7.12.2) ------------------

__device__ __forceinline__ v16h load_A(const _Float16* base, int stride, int kbase) {
    const int lane = threadIdx.x & 31;
    const int m = lane & 15, hi = lane >> 4;
    const _Float16* p = base + m * stride + kbase + hi * 8;
    union { v16h v; f4 f[2]; } u;
    u.f[0] = *(const f4*)p;        // k = kbase + hi*8 + [0..7]
    u.f[1] = *(const f4*)(p + 16); // k = kbase + hi*8 + 16 + [0..7]
    return u.v;
}

__device__ __forceinline__ v16h load_B(const _Float16* wT, int padK, int kbase) {
    const int lane = threadIdx.x & 31;
    const int n = lane & 15, hi = lane >> 4;
    const _Float16* p = wT + n * padK + kbase + hi * 8;
    union { v16h v; f4 f[2]; } u;
    u.f[0] = *(const f4*)p;        // k = kbase + hi*8 + [0..7]
    u.f[1] = *(const f4*)(p + 16); // k = kbase + hi*8 + 16 + [0..7]
    return u.v;
}

__device__ __forceinline__ void store_D(_Float16* base, int stride, int nbase,
                                        v8f d, bool relu) {
    const int lane = threadIdx.x & 31;
    const int n = lane & 15, hi = lane >> 4;
#pragma unroll
    for (int r = 0; r < 8; ++r) {
        float v = d[r];
        if (relu) v = fmaxf(v, 0.f);
        base[(r + hi * 8) * stride + nbase + n] = (_Float16)v;
    }
}

// ---- per-wave GEMM: [16 x K] @ [K x 16*NT] -> LDS ---------------------------

template <int NT>
__device__ __forceinline__ void wave_gemm(const _Float16* aBase, int sIn, int K,
                                          const _Float16* wT, int padK,
                                          _Float16* oBase, int sOut, bool relu) {
    v8f acc[NT];
#pragma unroll
    for (int i = 0; i < NT; ++i)
#pragma unroll
        for (int j = 0; j < 8; ++j) acc[i][j] = 0.f;

    for (int ks = 0; ks < K; ks += 32) {
        v16h a = load_A(aBase, sIn, ks);
#pragma unroll
        for (int nt = 0; nt < NT; ++nt) {
            v16h b = load_B(wT + nt * 16 * padK, padK, ks);
            acc[nt] = __builtin_amdgcn_wmma_f32_16x16x32_f16(
                false, a, false, b, (short)0, acc[nt], false, false);
        }
    }
#pragma unroll
    for (int nt = 0; nt < NT; ++nt) store_D(oBase, sOut, nt * 16, acc[nt], relu);
}

// MLP1 output layer fused with skip-blend: X = alpha*out1 + (1-alpha)*F
__device__ __forceinline__ void wave_gemm_blend(const _Float16* aBase, int sIn,
                                                const _Float16* wT, int padK,
                                                const _Float16* fBase, int sF,
                                                _Float16* oBase, int sOut,
                                                float alpha) {
    v8f acc[8];
#pragma unroll
    for (int i = 0; i < 8; ++i)
#pragma unroll
        for (int j = 0; j < 8; ++j) acc[i][j] = 0.f;

    for (int ks = 0; ks < 64; ks += 32) {
        v16h a = load_A(aBase, sIn, ks);
#pragma unroll
        for (int nt = 0; nt < 8; ++nt) {
            v16h b = load_B(wT + nt * 16 * padK, padK, ks);
            acc[nt] = __builtin_amdgcn_wmma_f32_16x16x32_f16(
                false, a, false, b, (short)0, acc[nt], false, false);
        }
    }
    const int lane = threadIdx.x & 31;
    const int n = lane & 15, hi = lane >> 4;
    const float om = 1.f - alpha;
#pragma unroll
    for (int nt = 0; nt < 8; ++nt)
#pragma unroll
        for (int r = 0; r < 8; ++r) {
            const int row = r + hi * 8, col = nt * 16 + n;
            float f = (float)fBase[row * sF + col];
            oBase[row * sOut + col] = (_Float16)(alpha * acc[nt][r] + om * f);
        }
}

// final layer: [16 x 64] @ [64 x 16] (only column 0 valid) -> global f32
__device__ __forceinline__ void wave_final(const _Float16* aBase, int sIn,
                                           const _Float16* wT, int padK,
                                           float* out) {
    v8f acc;
#pragma unroll
    for (int j = 0; j < 8; ++j) acc[j] = 0.f;
    for (int ks = 0; ks < 64; ks += 32) {
        v16h a = load_A(aBase, sIn, ks);
        v16h b = load_B(wT, padK, ks);
        acc = __builtin_amdgcn_wmma_f32_16x16x32_f16(
            false, a, false, b, (short)0, acc, false, false);
    }
    const int lane = threadIdx.x & 31;
    if ((lane & 15) == 0) {
        const int hi = lane >> 4;
#pragma unroll
        for (int r = 0; r < 8; ++r) out[hi * 8 + r] = acc[r];
    }
}

// ---- weight staging: global f32 [K x N] -> LDS f16 transposed [N x padK] ----

__device__ __forceinline__ void stage_wT(const float* __restrict__ W,
                                         int K, int N, int padK, _Float16* wT) {
    const int tot = K * N;
    for (int i = threadIdx.x; i < tot; i += 128) {
        const int k = i / N, n = i - k * N;
        wT[n * padK + k] = (_Float16)W[i]; // W[i] == W[k*N+n] (coalesced read)
    }
}

__device__ __forceinline__ void stage_w2out(const float* __restrict__ W,
                                            _Float16* wT) {
    // (64 x 1) weight padded to 16 output columns (cols 1..15 zero)
    for (int i = threadIdx.x; i < 16 * 64; i += 128) {
        const int n = i >> 6, k = i & 63;
        wT[n * 72 + k] = (n == 0) ? (_Float16)W[k] : (_Float16)0.f;
    }
}

// ---- hash-grid encoders (fp32 math, fp16 output into act buffer) ------------

__device__ __forceinline__ void encode3(const float* __restrict__ x,
                                        const float* __restrict__ tbl,
                                        _Float16* sAct, int stride, int pbase) {
    const int t = threadIdx.x;
    const int p = t >> 1;
    const int l0 = (t & 1) * 8;
    const float* xp = x + (size_t)(pbase + p) * 3;
    const float X = xp[0], Y = xp[1], Z = xp[2];
    _Float16* row = sAct + p * stride;
    for (int li = 0; li < 8; ++li) {
        const int l = l0 + li;
        const float res = c_res[l];
        const float px = X * res, py = Y * res, pz = Z * res;
        const float fx = floorf(px), fy = floorf(py), fz = floorf(pz);
        const unsigned x0 = (unsigned)fx, y0 = (unsigned)fy, z0 = (unsigned)fz;
        const float rx = px - fx, ry = py - fy, rz = pz - fz;
        const float* tl = tbl + (size_t)l * TBL_LEVEL_STRIDE;
        float a0 = 0.f, a1 = 0.f;
#pragma unroll
        for (int c = 0; c < 8; ++c) {
            const unsigned bx = c & 1u, by = (c >> 1) & 1u, bz = (c >> 2) & 1u;
            const unsigned h =
                (x0 + bx) ^ ((y0 + by) * 2654435761u) ^ ((z0 + bz) * 805459861u);
            const unsigned idx = h & HASH_T_MASK;
            const float w = (bx ? rx : 1.f - rx) * (by ? ry : 1.f - ry) *
                            (bz ? rz : 1.f - rz);
            const f2 f = *(const f2*)(tl + (size_t)idx * 2u);
            a0 += w * f.x;
            a1 += w * f.y;
        }
        row[l * 2 + 0] = (_Float16)a0;
        row[l * 2 + 1] = (_Float16)a1;
    }
}

__device__ __forceinline__ void encode4(const float* __restrict__ x, float tval,
                                        const float* __restrict__ tbl,
                                        _Float16* sAct, int stride, int pbase) {
    const int t = threadIdx.x;
    const int p = t >> 1;
    const int l0 = (t & 1) * 8;
    const float* xp = x + (size_t)(pbase + p) * 3;
    const float X = xp[0], Y = xp[1], Z = xp[2], W4 = tval;
    _Float16* row = sAct + p * stride;
    for (int li = 0; li < 8; ++li) {
        const int l = l0 + li;
        const float res = c_res[l];
        const float px = X * res, py = Y * res, pz = Z * res, pw = W4 * res;
        const float fx = floorf(px), fy = floorf(py), fz = floorf(pz),
                    fw = floorf(pw);
        const unsigned x0 = (unsigned)fx, y0 = (unsigned)fy, z0 = (unsigned)fz,
                       w0 = (unsigned)fw;
        const float rx = px - fx, ry = py - fy, rz = pz - fz, rw = pw - fw;
        const float* tl = tbl + (size_t)l * TBL_LEVEL_STRIDE;
        float a0 = 0.f, a1 = 0.f;
#pragma unroll
        for (int c = 0; c < 16; ++c) {
            const unsigned bx = c & 1u, by = (c >> 1) & 1u, bz = (c >> 2) & 1u,
                           bw = (c >> 3) & 1u;
            const unsigned h = (x0 + bx) ^ ((y0 + by) * 2654435761u) ^
                               ((z0 + bz) * 805459861u) ^
                               ((w0 + bw) * 3674653429u);
            const unsigned idx = h & HASH_T_MASK;
            const float w = (bx ? rx : 1.f - rx) * (by ? ry : 1.f - ry) *
                            (bz ? rz : 1.f - rz) * (bw ? rw : 1.f - rw);
            const f2 f = *(const f2*)(tl + (size_t)idx * 2u);
            a0 += w * f.x;
            a1 += w * f.y;
        }
        row[l * 2 + 0] = (_Float16)a0;
        row[l * 2 + 1] = (_Float16)a1;
    }
}

// ---- fused kernel -----------------------------------------------------------

#define STAGE(Wp, K, N, PAD)                     \
    do {                                         \
        __syncthreads();                         \
        stage_wT((Wp), (K), (N), (PAD), sW);     \
        __syncthreads();                         \
    } while (0)

__global__ __launch_bounds__(128, 1) void fused_hash_mlp_kernel(
    const float* __restrict__ x, const float* __restrict__ tscalar,
    const float* __restrict__ alpha_p, const float* __restrict__ table_s,
    const float* __restrict__ ws_in, const float* __restrict__ ws_hid,
    const float* __restrict__ ws_out, const float* __restrict__ table_d,
    const float* __restrict__ wd_in, const float* __restrict__ wd_hid,
    const float* __restrict__ wd_out, const float* __restrict__ w1_in,
    const float* __restrict__ w1_hid, const float* __restrict__ w1_out,
    const float* __restrict__ w2_in, const float* __restrict__ w2_hid,
    const float* __restrict__ w2_out, float* __restrict__ out) {
    // 64x136 padded act buffers: 68-dword row stride -> conflict-free b128
    __shared__ __align__(16) _Float16 sF[64 * 136]; // features / blend source
    __shared__ __align__(16) _Float16 sX[64 * 136]; // act ping (128 wide)
    __shared__ __align__(16) _Float16 sY[64 * 72];  // act pong (64 wide)
    __shared__ __align__(16) _Float16 sW[9216];     // staged W^T (max 128x72)

    const int wave = threadIdx.x >> 5;
    const int pbase = blockIdx.x * 64;
    _Float16* aF = sF + wave * 16 * 136;
    _Float16* aX = sX + wave * 16 * 136;
    _Float16* aY = sY + wave * 16 * 72;

    const float tval = tscalar[0];
    const float alpha = alpha_p[0];

    // ---- static branch: encode -> 32-64-64-64-64-64 MLP -> sF[:, 0:64]
    encode3(x, table_s, sX, 136, pbase);
    STAGE(ws_in, 32, 64, 40);
    wave_gemm<4>(aX, 136, 32, sW, 40, aY, 72, true);
    STAGE(ws_hid + 0 * 4096, 64, 64, 72);
    wave_gemm<4>(aY, 72, 64, sW, 72, aX, 136, true);
    STAGE(ws_hid + 1 * 4096, 64, 64, 72);
    wave_gemm<4>(aX, 136, 64, sW, 72, aY, 72, true);
    STAGE(ws_hid + 2 * 4096, 64, 64, 72);
    wave_gemm<4>(aY, 72, 64, sW, 72, aX, 136, true);
    STAGE(ws_out, 64, 64, 72);
    wave_gemm<4>(aX, 136, 64, sW, 72, aF, 136, false); // linear out

    // ---- dynamic branch: encode4 -> MLP -> sF[:, 64:128]
    encode4(x, tval, table_d, sX, 136, pbase);
    STAGE(wd_in, 32, 64, 40);
    wave_gemm<4>(aX, 136, 32, sW, 40, aY, 72, true);
    STAGE(wd_hid + 0 * 4096, 64, 64, 72);
    wave_gemm<4>(aY, 72, 64, sW, 72, aX, 136, true);
    STAGE(wd_hid + 1 * 4096, 64, 64, 72);
    wave_gemm<4>(aX, 136, 64, sW, 72, aY, 72, true);
    STAGE(wd_hid + 2 * 4096, 64, 64, 72);
    wave_gemm<4>(aY, 72, 64, sW, 72, aX, 136, true);
    STAGE(wd_out, 64, 64, 72);
    wave_gemm<4>(aX, 136, 64, sW, 72, aF + 64, 136, false); // linear out

    // ---- MLP1 (128-64-64-128) + skip blend -> sX (128 wide)
    STAGE(w1_in, 128, 64, 136);
    wave_gemm<4>(aF, 136, 128, sW, 136, aY, 72, true);
    STAGE(w1_hid, 64, 64, 72);
    wave_gemm<4>(aY, 72, 64, sW, 72, aX, 136, true);
    STAGE(w1_out, 64, 128, 72);
    wave_gemm_blend(aX, 136, sW, 72, aF, 136, aX, 136, alpha);

    // ---- MLP2 (128-64-64-1) -> global
    STAGE(w2_in, 128, 64, 136);
    wave_gemm<4>(aX, 136, 128, sW, 136, aY, 72, true);
    STAGE(w2_hid, 64, 64, 72);
    wave_gemm<4>(aY, 72, 64, sW, 72, aX, 136, true);
    __syncthreads();
    stage_w2out(w2_out, sW);
    __syncthreads();
    wave_final(aX, 136, sW, 72, out + pbase + wave * 16);
}

// ---------------------------------------------------------------------------

extern "C" void kernel_launch(void* const* d_in, const int* in_sizes, int n_in,
                              void* d_out, int out_size, void* d_ws,
                              size_t ws_size, hipStream_t stream) {
    const float* x       = (const float*)d_in[0];
    const float* t       = (const float*)d_in[1];
    const float* alpha   = (const float*)d_in[2];
    const float* table_s = (const float*)d_in[3];
    const float* ws_in   = (const float*)d_in[4];
    const float* ws_hid  = (const float*)d_in[5];
    const float* ws_out  = (const float*)d_in[6];
    const float* table_d = (const float*)d_in[7];
    const float* wd_in   = (const float*)d_in[8];
    const float* wd_hid  = (const float*)d_in[9];
    const float* wd_out  = (const float*)d_in[10];
    const float* w1_in   = (const float*)d_in[11];
    const float* w1_hid  = (const float*)d_in[12];
    const float* w1_out  = (const float*)d_in[13];
    const float* w2_in   = (const float*)d_in[14];
    const float* w2_hid  = (const float*)d_in[15];
    const float* w2_out  = (const float*)d_in[16];

    const int n = in_sizes[0] / 3;      // 524288 points
    const int blocks = n / 64;          // 64 points per block

    fused_hash_mlp_kernel<<<dim3(blocks), dim3(128), 0, stream>>>(
        x, t, alpha, table_s, ws_in, ws_hid, ws_out, table_d, wd_in, wd_hid,
        wd_out, w1_in, w1_hid, w1_out, w2_in, w2_hid, w2_out, (float*)d_out);
}
